// VanillaRNN_10041633538495
// MI455X (gfx1250) — matrix operand order repaired and checked by reading, not compiled, not run on hardware
//
#include <hip/hip_runtime.h>
#include <cmath>

typedef __attribute__((ext_vector_type(16))) _Float16 v16h;
typedef __attribute__((ext_vector_type(8)))  float    v8f;
typedef int v4i __attribute__((vector_size(16)));   // matches builtin's expected pointee

#define B_DIM 1024
#define S_DIM 128
#define D_DIM 256
#define H_DIM 2048
#define O_DIM 10
#define V_DIM 10

#define BM 128
#define BN 64
#define BK 64
#define LDSS 72   // padded LDS row stride (halves): 144B = 9*16B, keeps 16B alignment

// ---- gfx1250 async memory->LDS copy (ASYNCcnt path), guarded for portability ----
#if defined(__has_builtin)
#  if __has_builtin(__builtin_amdgcn_global_load_async_to_lds_b128)
#    define USE_ASYNC_COPY 1
#  endif
#endif

__device__ __forceinline__ void wait_async0() {
#if defined(__has_builtin) && __has_builtin(__builtin_amdgcn_s_wait_asynccnt)
  __builtin_amdgcn_s_wait_asynccnt(0);
#else
  asm volatile("s_wait_asynccnt 0x0" ::: "memory");
#endif
}

#ifdef USE_ASYNC_COPY
#define ASYNC_CP16(gsrc, ldst)                                          \
  __builtin_amdgcn_global_load_async_to_lds_b128(                       \
      (__attribute__((address_space(1))) v4i*)(void*)(gsrc),            \
      (__attribute__((address_space(3))) v4i*)(void*)(ldst), 0, 0)
#endif

// branch-free tanh: clamp + single v_exp_f32; error << f16 storage quantization
__device__ __forceinline__ float fast_tanh(float x) {
  float xc = fminf(fmaxf(x, -15.0f), 15.0f);
  float e  = __expf(2.0f * xc);
  return __fdividef(e - 1.0f, e + 1.0f);
}

// ---------------- table[v][h] = embed[v] . W_hx[h] + b_hx[h] ----------------
__global__ void table_kernel(const float* __restrict__ embed,
                             const float* __restrict__ W_hx,
                             const float* __restrict__ b_hx,
                             float* __restrict__ table) {
  int id = blockIdx.x * blockDim.x + threadIdx.x;
  if (id >= V_DIM * H_DIM) return;
  int v = id / H_DIM, h = id % H_DIM;
  const float* e = embed + v * D_DIM;
  const float* w = W_hx + h * D_DIM;
  float acc = b_hx[h];
#pragma unroll 4
  for (int d = 0; d < D_DIM; ++d) acc += e[d] * w[d];
  table[id] = acc;
}

// ---------------- fp32 -> f16 convert (W_hh) ----------------
__global__ void cvt_kernel(const float* __restrict__ src,
                           _Float16* __restrict__ dst, int n) {
  int id = blockIdx.x * blockDim.x + threadIdx.x;
  if (id < n) dst[id] = (_Float16)src[id];
}

// ---------------- h0 = tanh(table[x[b,0]]) ----------------
__global__ void h0_kernel(const int* __restrict__ x,
                          const float* __restrict__ table,
                          _Float16* __restrict__ h) {
  int id = blockIdx.x * blockDim.x + threadIdx.x;   // b*H + n
  int b = id >> 11, n = id & (H_DIM - 1);
  int idx = x[b * S_DIM];
  h[id] = (_Float16)fast_tanh(table[idx * H_DIM + n]);
}

// ---------------- one recurrence step: hout = tanh(table[x[:,t]] + hin @ Whhᵀ + b_hh) ----------------
__global__ __launch_bounds__(256)
void rnn_step_kernel(const _Float16* __restrict__ hin,
                     _Float16* __restrict__ hout,
                     const _Float16* __restrict__ W16,
                     const float* __restrict__ table,
                     const float* __restrict__ b_hh,
                     const int* __restrict__ x, int t) {
  __shared__ __align__(16) _Float16 As[BM][LDSS];
  __shared__ __align__(16) _Float16 Bs[BN][LDSS];

  const int tid   = threadIdx.x;
  const int lane  = tid & 31;
  const int wave  = tid >> 5;
  const int mbase = blockIdx.y * BM;
  const int nbase = blockIdx.x * BN;
  const int wm    = (wave >> 1) * 32;   // 4 waves along M
  const int wn    = (wave & 1) * 32;    // 2 waves along N
  const int l15   = lane & 15;
  const int khalf = lane >> 4;

  // staging geometry: rows of 64 halves = 8 chunks of 8 halves (16B)
  const int srow  = tid >> 3;           // 0..31
  const int spart = (tid & 7) * 8;      // column offset in halves

  v8f zero = {};
  v8f acc[2][2];
  acc[0][0] = zero; acc[0][1] = zero; acc[1][0] = zero; acc[1][1] = zero;

  for (int k0 = 0; k0 < H_DIM; k0 += BK) {
    // --- stage A tile (128x64 halves: 4 chunks/thread) + B tile (64x64: 2 chunks/thread) ---
#ifdef USE_ASYNC_COPY
#pragma unroll
    for (int i = 0; i < 4; ++i) {
      const int row = srow + 32 * i;
      ASYNC_CP16(hin + (mbase + row) * H_DIM + k0 + spart, &As[row][spart]);
    }
#pragma unroll
    for (int i = 0; i < 2; ++i) {
      const int row = srow + 32 * i;
      ASYNC_CP16(W16 + (nbase + row) * H_DIM + k0 + spart, &Bs[row][spart]);
    }
    if (k0 + BK < H_DIM)
      __builtin_prefetch(W16 + (nbase + srow) * H_DIM + k0 + BK, 0, 1);
    wait_async0();
#else
#pragma unroll
    for (int i = 0; i < 4; ++i) {
      const int row = srow + 32 * i;
      *reinterpret_cast<uint4*>(&As[row][spart]) =
          *reinterpret_cast<const uint4*>(hin + (mbase + row) * H_DIM + k0 + spart);
    }
#pragma unroll
    for (int i = 0; i < 2; ++i) {
      const int row = srow + 32 * i;
      *reinterpret_cast<uint4*>(&Bs[row][spart]) =
          *reinterpret_cast<const uint4*>(W16 + (nbase + row) * H_DIM + k0 + spart);
    }
    if (k0 + BK < H_DIM)
      __builtin_prefetch(W16 + (nbase + srow) * H_DIM + k0 + BK, 0, 1);
#endif
    __syncthreads();

    // --- two K=32 sub-chunks per stage: 8 WMMAs per barrier pair ---
#pragma unroll
    for (int kk = 0; kk < BK; kk += 32) {
      v16h afrag[2], bfrag[2];
#pragma unroll
      for (int mi = 0; mi < 2; ++mi) {       // A 16x32 f16 layout (ISA table)
        union { v16h v; uint4 q[2]; } u;
        const int m  = wm + mi * 16 + l15;
        const int kb = kk + khalf * 8;       // lanes 0-15: K=0..7,16..23; 16-31: +8
        u.q[0] = *reinterpret_cast<const uint4*>(&As[m][kb]);
        u.q[1] = *reinterpret_cast<const uint4*>(&As[m][kb + 16]);
        afrag[mi] = u.v;
      }
#pragma unroll
      for (int ni = 0; ni < 2; ++ni) {       // B 32x16 f16 layout
        union { v16h v; uint4 q[2]; } u;
        const int n  = wn + ni * 16 + l15;
        const int kb = kk + khalf * 8;       // lanes 0-15: K=0..7,16..23; 16-31: +8
        u.q[0] = *reinterpret_cast<const uint4*>(&Bs[n][kb]);
        u.q[1] = *reinterpret_cast<const uint4*>(&Bs[n][kb + 16]);
        bfrag[ni] = u.v;
      }
#pragma unroll
      for (int mi = 0; mi < 2; ++mi)
#pragma unroll
        for (int ni = 0; ni < 2; ++ni)
          acc[mi][ni] = __builtin_amdgcn_wmma_f32_16x16x32_f16(
              false, afrag[mi], false, bfrag[ni], (short)0, acc[mi][ni], false, false);
    }
    __syncthreads();
  }

  // --- epilogue: + table[x[b,t]] + b_hh, tanh, store f16 ---
#pragma unroll
  for (int mi = 0; mi < 2; ++mi) {
#pragma unroll
    for (int ni = 0; ni < 2; ++ni) {
      const int n_g  = nbase + wn + ni * 16 + l15;
      const float bb = b_hh[n_g];
#pragma unroll
      for (int r = 0; r < 8; ++r) {
        const int m_g = mbase + wm + mi * 16 + r + 8 * khalf;
        const int idx = x[m_g * S_DIM + t];
        float val = acc[mi][ni][r] + table[idx * H_DIM + n_g] + bb;
        hout[m_g * H_DIM + n_g] = (_Float16)fast_tanh(val);
      }
    }
  }
}

// ---------------- output projection (Hx10) + softmax ----------------
__global__ __launch_bounds__(256)
void out_kernel(const _Float16* __restrict__ hT,
                const float* __restrict__ W_oh,
                const float* __restrict__ b_oh,
                float* __restrict__ out) {
  __shared__ float red[O_DIM][256];
  const int b = blockIdx.x, tid = threadIdx.x;
  float acc[O_DIM];
#pragma unroll
  for (int j = 0; j < O_DIM; ++j) acc[j] = 0.0f;
  for (int h = tid; h < H_DIM; h += 256) {
    float hv = (float)hT[b * H_DIM + h];
#pragma unroll
    for (int j = 0; j < O_DIM; ++j) acc[j] += hv * W_oh[j * H_DIM + h];
  }
#pragma unroll
  for (int j = 0; j < O_DIM; ++j) red[j][tid] = acc[j];
  __syncthreads();
  for (int s = 128; s > 0; s >>= 1) {
    if (tid < s)
#pragma unroll
      for (int j = 0; j < O_DIM; ++j) red[j][tid] += red[j][tid + s];
    __syncthreads();
  }
  if (tid == 0) {
    float o[O_DIM], mx = -1e30f, sum = 0.0f;
#pragma unroll
    for (int j = 0; j < O_DIM; ++j) { o[j] = red[j][0] + b_oh[j]; mx = fmaxf(mx, o[j]); }
#pragma unroll
    for (int j = 0; j < O_DIM; ++j) { o[j] = __expf(o[j] - mx); sum += o[j]; }
#pragma unroll
    for (int j = 0; j < O_DIM; ++j) out[b * O_DIM + j] = o[j] / sum;
  }
}

extern "C" void kernel_launch(void* const* d_in, const int* in_sizes, int n_in,
                              void* d_out, int out_size, void* d_ws, size_t ws_size,
                              hipStream_t stream) {
  const int*   x     = (const int*)d_in[0];
  const float* embed = (const float*)d_in[1];
  const float* W_hx  = (const float*)d_in[2];
  const float* b_hx  = (const float*)d_in[3];
  const float* W_hh  = (const float*)d_in[4];
  const float* b_hh  = (const float*)d_in[5];
  const float* W_oh  = (const float*)d_in[6];
  const float* b_oh  = (const float*)d_in[7];
  float* out = (float*)d_out;
  (void)in_sizes; (void)n_in; (void)out_size; (void)ws_size;

  char* ws = (char*)d_ws;
  float*    table = (float*)(ws + 0);                 //    80 KB
  _Float16* W16   = (_Float16*)(ws + 81920);          //   8 MB
  _Float16* hA    = (_Float16*)(ws + 8470528);        //   4 MB
  _Float16* hB    = (_Float16*)(ws + 12664832);       //   4 MB (total ~16.9 MB)

  table_kernel<<<(V_DIM * H_DIM + 255) / 256, 256, 0, stream>>>(embed, W_hx, b_hx, table);
  cvt_kernel<<<(H_DIM * H_DIM) / 256, 256, 0, stream>>>(W_hh, W16, H_DIM * H_DIM);
  h0_kernel<<<(B_DIM * H_DIM) / 256, 256, 0, stream>>>(x, table, hA);

  dim3 grid(H_DIM / BN, B_DIM / BM);   // (32, 8) blocks of 256 threads
  _Float16 *src = hA, *dst = hB;
  for (int t = 1; t < S_DIM; ++t) {
    rnn_step_kernel<<<grid, 256, 0, stream>>>(src, dst, W16, table, b_hh, x, t);
    _Float16* tmp = src; src = dst; dst = tmp;
  }

  out_kernel<<<B_DIM, 256, 0, stream>>>(src, W_oh, b_oh, out);
}
